// DLSMNLayer_8529805050132
// MI455X (gfx1250) — matrix operand order repaired and checked by reading, not compiled, not run on hardware
//
#include <hip/hip_runtime.h>
#include <hip/hip_bf16.h>

// ---------------------------------------------------------------------------
// Problem constants (from reference): B=4, S=4096, D=1024, H=16, HD=64,
// DC=512, K=64, L=12, LAYER=3, T=256, TEMP=1, EPS=1e-6.
// ---------------------------------------------------------------------------
static constexpr int  Bh   = 4;
static constexpr int  Sh   = 4096;
static constexpr int  Dh   = 1024;
static constexpr int  Hh   = 16;
static constexpr int  DCh  = 512;
static constexpr int  Kslt = 64;
static constexpr int  Lh   = 12;
static constexpr int  LAY  = 3;
static constexpr int  Th   = 256;
static constexpr int  MS   = Lh * Kslt;            // 768 cache slots
static constexpr long long BSD = (long long)Bh * Sh * Dh;   // 16,777,216

// ---------------------------------------------------------------------------
// bf16 helpers
// ---------------------------------------------------------------------------
typedef __attribute__((ext_vector_type(16))) __bf16 v16bf;
typedef __attribute__((ext_vector_type(8)))  __bf16 v8bf;
typedef __attribute__((ext_vector_type(4)))  __bf16 v4bf;
typedef __attribute__((ext_vector_type(8)))  float  v8f;

struct BF16x16 { v8bf lo, hi; };

__device__ __forceinline__ __bf16 f2bf(float f) {
  unsigned u = __builtin_bit_cast(unsigned, f);
  unsigned r = u + 0x7FFFu + ((u >> 16) & 1u);
  unsigned short h = (unsigned short)(r >> 16);
  return __builtin_bit_cast(__bf16, h);
}
__device__ __forceinline__ float bf2f(__bf16 b) {
  unsigned short h = __builtin_bit_cast(unsigned short, b);
  unsigned u = ((unsigned)h) << 16;
  return __builtin_bit_cast(float, u);
}

// ---------------------------------------------------------------------------
// CDNA5 async global->LDS copy (ASYNCcnt tracked).  Per-lane 16B transfer:
//   global_load_async_to_lds_b128 vdst_lds_byte_off, v[addr:addr+1], off
// ---------------------------------------------------------------------------
__device__ __forceinline__ void async_copy_b128(unsigned lds_byte_off, const void* g) {
  asm volatile("global_load_async_to_lds_b128 %0, %1, off"
               :: "v"(lds_byte_off), "v"(g)
               : "memory");
}
__device__ __forceinline__ void wait_async0() {
  asm volatile("s_wait_asynccnt 0x0" ::: "memory");
}

// ---------------------------------------------------------------------------
// Generic bf16 WMMA GEMM: C[z] = epi( scale * A[z] @ B[z] )
//   block = 256 threads = 8 wave32s; tile 128(M) x 128(N) x 32(K)
//   waves 4(M) x 2(N); each wave owns 32x64 = 2x4 WMMA 16x16x32 tiles.
//   A tile in LDS row-major [m][k] (fragment = 2 aligned ds_load_b128);
//   B tile in LDS transposed [n][k] (fragment = 2 aligned ds_load_b128).
// EPI: 0=scale 1=+bias 2=gelu(+bias) 3=elu+1 4=+add 5=add+gate[m]*acc 6=+bias+add
// ---------------------------------------------------------------------------
template <bool TA, bool TB, bool ABF, bool CBF, int EPI>
__global__ __launch_bounds__(256)
void gemm_wmma(const void* __restrict__ Av, int lda, long long sAz,
               const float* __restrict__ Bm, int ldb, long long sBz,
               void* __restrict__ Cv, int ldc, long long sCz,
               int M, int N, int Kd,
               const float* __restrict__ bias,
               const float* __restrict__ gate, long long sGz,
               const float* __restrict__ add,
               float scale)
{
  constexpr int BM = 128, BN = 128, BK = 32, BKP = BK + 8;  // 80B rows, 16B aligned
  const int z = blockIdx.z;
  const float*  Af = (const float*)Av;
  const __bf16* Ab = (const __bf16*)Av;
  float*  Cf = (float*)Cv;
  __bf16* Cb = (__bf16*)Cv;

  const int tid  = threadIdx.x;
  const int lane = tid & 31;
  const int w    = tid >> 5;
  const int wm   = w >> 1;        // 0..3
  const int wn   = w & 1;         // 0..1
  const int m0   = blockIdx.y * BM;
  const int n0   = blockIdx.x * BN;

  __shared__ __bf16 As[BM][BKP];  // [m][k]
  __shared__ __bf16 Bt[BN][BKP];  // [n][k]  (transposed B)

  v8f acc[2][4];
  #pragma unroll
  for (int mi = 0; mi < 2; ++mi)
    #pragma unroll
    for (int ni = 0; ni < 4; ++ni)
      #pragma unroll
      for (int r = 0; r < 8; ++r) acc[mi][ni][r] = 0.0f;

  const __bf16 bz = __builtin_bit_cast(__bf16, (unsigned short)0);
  const bool fullM = (m0 + BM <= M);
  const bool fullN = (n0 + BN <= N);

  for (int k0 = 0; k0 < Kd; k0 += BK) {
    const bool fullK = (k0 + BK <= Kd);

    // ---- stage A tile: 1024 chunks of 8 elems; 4 chunks per thread ----
    #pragma unroll
    for (int c = 0; c < 4; ++c) {
      int id = tid + 256 * c;
      int mm = id >> 2;
      int kc = (id & 3) * 8;
      int gm = m0 + mm, gk0 = k0 + kc;
      v8bf a8;
      if (ABF) {
        const __bf16* src = Ab + (long long)z * sAz + (long long)gm * lda + gk0;
        if (fullM && fullK) {
          a8 = *(const v8bf*)src;
        } else {
          #pragma unroll
          for (int j = 0; j < 8; ++j)
            a8[j] = (gm < M && (gk0 + j) < Kd) ? src[j] : bz;
        }
      } else if (!TA) {
        const float* src = Af + (long long)z * sAz + (long long)gm * lda + gk0;
        if (fullM && fullK && ((lda & 3) == 0)) {
          float4 f0 = *(const float4*)(src);
          float4 f1 = *(const float4*)(src + 4);
          a8[0] = f2bf(f0.x); a8[1] = f2bf(f0.y); a8[2] = f2bf(f0.z); a8[3] = f2bf(f0.w);
          a8[4] = f2bf(f1.x); a8[5] = f2bf(f1.y); a8[6] = f2bf(f1.z); a8[7] = f2bf(f1.w);
        } else {
          #pragma unroll
          for (int j = 0; j < 8; ++j)
            a8[j] = (gm < M && (gk0 + j) < Kd) ? f2bf(src[j]) : bz;
        }
      } else {  // TA: strided gather
        #pragma unroll
        for (int j = 0; j < 8; ++j) {
          int gk = gk0 + j;
          a8[j] = (gm < M && gk < Kd)
                    ? f2bf(Af[(long long)z * sAz + (long long)gk * lda + gm]) : bz;
        }
      }
      *(v8bf*)&As[mm][kc] = a8;
    }

    // ---- stage B tile (transposed into Bt[n][k]) ----
    #pragma unroll
    for (int c = 0; c < 4; ++c) {
      int id = tid + 256 * c;
      int nn = id >> 2;
      int kc = (id & 3) * 8;
      int gn = n0 + nn, gk0 = k0 + kc;
      v8bf b8;
      if (TB) {
        const float* src = Bm + (long long)z * sBz + (long long)gn * ldb + gk0;
        if (fullN && fullK && ((ldb & 3) == 0)) {
          float4 f0 = *(const float4*)(src);
          float4 f1 = *(const float4*)(src + 4);
          b8[0] = f2bf(f0.x); b8[1] = f2bf(f0.y); b8[2] = f2bf(f0.z); b8[3] = f2bf(f0.w);
          b8[4] = f2bf(f1.x); b8[5] = f2bf(f1.y); b8[6] = f2bf(f1.z); b8[7] = f2bf(f1.w);
        } else {
          #pragma unroll
          for (int j = 0; j < 8; ++j)
            b8[j] = (gn < N && (gk0 + j) < Kd) ? f2bf(src[j]) : bz;
        }
      } else {  // row-major weights: strided gather while transposing
        #pragma unroll
        for (int j = 0; j < 8; ++j) {
          int gk = gk0 + j;
          b8[j] = (gn < N && gk < Kd)
                    ? f2bf(Bm[(long long)z * sBz + (long long)gk * ldb + gn]) : bz;
        }
      }
      *(v8bf*)&Bt[nn][kc] = b8;
    }

    // prefetch next B tile rows into cache (global_prefetch_b8)
    if (!TB && (k0 + BK) < Kd) {
      int kk = tid >> 3, nn = (tid & 7) * 16;
      int gk = k0 + BK + kk, gn = n0 + nn;
      if (gk < Kd && gn < N)
        __builtin_prefetch(&Bm[(long long)z * sBz + (long long)gk * ldb + gn], 0, 1);
    }
    __syncthreads();

    // ---- fragments: aligned 16B LDS vector loads ----
    const int kg = lane >> 4, l15 = lane & 15;
    v16bf af[2];
    #pragma unroll
    for (int mi = 0; mi < 2; ++mi) {
      int row = wm * 32 + mi * 16 + l15;
      v8bf lo = *(const v8bf*)&As[row][kg * 8];
      v8bf hi = *(const v8bf*)&As[row][16 + kg * 8];
      af[mi] = __builtin_bit_cast(v16bf, BF16x16{lo, hi});
    }
    v16bf bf4[4];
    #pragma unroll
    for (int ni = 0; ni < 4; ++ni) {
      int col = wn * 64 + ni * 16 + l15;
      v8bf lo = *(const v8bf*)&Bt[col][kg * 8];
      v8bf hi = *(const v8bf*)&Bt[col][16 + kg * 8];
      bf4[ni] = __builtin_bit_cast(v16bf, BF16x16{lo, hi});
    }
    #pragma unroll
    for (int mi = 0; mi < 2; ++mi)
      #pragma unroll
      for (int ni = 0; ni < 4; ++ni)
        acc[mi][ni] = __builtin_amdgcn_wmma_f32_16x16x32_bf16(
            false, af[mi], false, bf4[ni], (short)0, acc[mi][ni], false, false);
    __syncthreads();
  }

  // ---- epilogue + store ----
  #pragma unroll
  for (int mi = 0; mi < 2; ++mi) {
    #pragma unroll
    for (int ni = 0; ni < 4; ++ni) {
      int n = n0 + wn * 64 + ni * 16 + (lane & 15);
      #pragma unroll
      for (int r = 0; r < 8; ++r) {
        int m = m0 + wm * 32 + mi * 16 + (lane >> 4) * 8 + r;
        if (m < M && n < N) {
          long long cidx = (long long)z * sCz + (long long)m * ldc + n;
          float v = acc[mi][ni][r] * scale;
          if (EPI == 1) { v += bias[n]; }
          else if (EPI == 2) { v += bias[n]; v = 0.5f * v * (1.0f + erff(v * 0.70710678118f)); }
          else if (EPI == 3) { v = (v > 0.0f) ? (v + 1.0f) : __expf(v); }
          else if (EPI == 4) { v += add[cidx]; }
          else if (EPI == 5) { v = add[cidx] + gate[(long long)z * sGz + m] * v; }
          else if (EPI == 6) { v += bias[n] + add[cidx]; }
          if (CBF) Cb[cidx] = f2bf(v); else Cf[cidx] = v;
        }
      }
    }
  }
}

// ---------------------------------------------------------------------------
// Row softmax (in-place, one block per row)
// ---------------------------------------------------------------------------
__global__ __launch_bounds__(256)
void softmax_rows(float* __restrict__ data, int ncol)
{
  long long row = blockIdx.x;
  float* p = data + row * (long long)ncol;
  __shared__ float red[256];
  int tid = threadIdx.x;
  float mx = -1e30f;
  for (int i = tid; i < ncol; i += 256) mx = fmaxf(mx, p[i]);
  red[tid] = mx; __syncthreads();
  for (int s = 128; s > 0; s >>= 1) { if (tid < s) red[tid] = fmaxf(red[tid], red[tid + s]); __syncthreads(); }
  mx = red[0]; __syncthreads();
  float sum = 0.0f;
  for (int i = tid; i < ncol; i += 256) { float e = __expf(p[i] - mx); p[i] = e; sum += e; }
  red[tid] = sum; __syncthreads();
  for (int s = 128; s > 0; s >>= 1) { if (tid < s) red[tid] += red[tid + s]; __syncthreads(); }
  float inv = 1.0f / red[0];
  for (int i = tid; i < ncol; i += 256) p[i] *= inv;
}

// ---------------------------------------------------------------------------
// sigmoid(row . w + bias): one wave32 per row, 8 rows per block
// ---------------------------------------------------------------------------
__global__ __launch_bounds__(256)
void rowdot_sigmoid(const float* __restrict__ in, long long sAz, int rowsPerZ, int Kd,
                    const float* __restrict__ wv, const float* __restrict__ bias,
                    float* __restrict__ out, long long sOz)
{
  int z = blockIdx.y;
  int w = threadIdx.x >> 5, lane = threadIdx.x & 31;
  int row = blockIdx.x * 8 + w;
  if (row >= rowsPerZ) return;
  const float* p = in + (long long)z * sAz + (long long)row * Kd;
  float s = 0.0f;
  for (int i = lane; i < Kd; i += 32) s += p[i] * wv[i];
  #pragma unroll
  for (int m = 16; m > 0; m >>= 1) s += __shfl_xor(s, m, 32);
  if (lane == 0) out[(long long)z * sOz + row] = 1.0f / (1.0f + __expf(-(s + bias[0])));
}

// ---------------------------------------------------------------------------
// Linear attention kv[b,h] = pk^T @ v (64x64, K-reduce over S).
// fp32 tiles staged via async global->LDS (ASYNCcnt), then converted to
// transposed bf16 tiles for aligned ds_load_b128 fragment reads.
// ---------------------------------------------------------------------------
__global__ __launch_bounds__(256)
void linattn_kv(const float* __restrict__ pk, const float* __restrict__ v,
                float* __restrict__ kv)
{
  const int bh = blockIdx.x;
  const int b = bh >> 4, h = bh & 15;
  const int tid = threadIdx.x, lane = tid & 31, w = tid >> 5;
  __shared__ float  stage[2][32 * 64];   // raw fp32 async landing zone (16KB)
  __shared__ __bf16 pkt[64][40];         // pk^T : [d][s]
  __shared__ __bf16 vt[64][40];          // v^T  : [e][s]
  v8f acc[2];
  #pragma unroll
  for (int ni = 0; ni < 2; ++ni)
    #pragma unroll
    for (int r = 0; r < 8; ++r) acc[ni][r] = 0.0f;
  const long long base = ((long long)b * Sh) * Dh + h * 64;

  for (int s0 = 0; s0 < Sh; s0 += 32) {
    // async-copy 32x64 fp32 of pk and v (16B per lane per issue)
    #pragma unroll
    for (int c = 0; c < 2; ++c) {
      int id = tid + 256 * c;            // 0..511 chunks of 4 floats
      int s  = id >> 4;
      int dc = (id & 15) << 2;
      long long g = base + (long long)(s0 + s) * Dh + dc;
      async_copy_b128((unsigned)(uintptr_t)&stage[0][id * 4], &pk[g]);
      async_copy_b128((unsigned)(uintptr_t)&stage[1][id * 4], &v[g]);
    }
    wait_async0();
    __syncthreads();
    // convert + transpose into bf16 tiles
    for (int i = tid; i < 32 * 64; i += 256) {
      int s = i >> 6, d = i & 63;
      pkt[d][s] = f2bf(stage[0][i]);
      vt[d][s]  = f2bf(stage[1][i]);
    }
    __syncthreads();

    const int kg = lane >> 4, l15 = lane & 15;
    int rowd = (w >> 1) * 16 + l15;      // A = pk^T: row index = d
    v8bf alo = *(const v8bf*)&pkt[rowd][kg * 8];
    v8bf ahi = *(const v8bf*)&pkt[rowd][16 + kg * 8];
    v16bf a = __builtin_bit_cast(v16bf, BF16x16{alo, ahi});
    #pragma unroll
    for (int ni = 0; ni < 2; ++ni) {
      int col = (w & 1) * 32 + ni * 16 + l15;
      v8bf blo = *(const v8bf*)&vt[col][kg * 8];
      v8bf bhi = *(const v8bf*)&vt[col][16 + kg * 8];
      v16bf bb = __builtin_bit_cast(v16bf, BF16x16{blo, bhi});
      acc[ni] = __builtin_amdgcn_wmma_f32_16x16x32_bf16(
          false, a, false, bb, (short)0, acc[ni], false, false);
    }
    __syncthreads();
  }
  #pragma unroll
  for (int ni = 0; ni < 2; ++ni) {
    int n = (w & 1) * 32 + ni * 16 + (lane & 15);
    #pragma unroll
    for (int r = 0; r < 8; ++r) {
      int m = (w >> 1) * 16 + (lane >> 4) * 8 + r;
      kv[(long long)bh * 4096 + m * 64 + n] = acc[ni][r];
    }
  }
}

// ksum[b,h,d] = sum_s pk[b,h,s,d]   grid=B*H, block=64
__global__ __launch_bounds__(64)
void ksum_kernel(const float* __restrict__ pk, float* __restrict__ ksum)
{
  int bh = blockIdx.x; int b = bh >> 4, h = bh & 15, d = threadIdx.x;
  long long base = ((long long)b * Sh) * Dh + h * 64 + d;
  float s = 0.0f;
  for (int i = 0; i < Sh; ++i) s += pk[base + (long long)i * Dh];
  ksum[bh * 64 + d] = s;
}

// num/den: out[b,s,h*64+e] = (pq @ kv) / (pq . ksum + eps)
// grid = (S/128, B*H); wave w owns rows w*16..+16, all 64 cols.
__global__ __launch_bounds__(256)
void linattn_num(const float* __restrict__ pq, const float* __restrict__ kv,
                 const float* __restrict__ ksum, float* __restrict__ out)
{
  const int bh = blockIdx.y;
  const int b = bh >> 4, h = bh & 15;
  const int s0 = blockIdx.x * 128;
  const int tid = threadIdx.x, lane = tid & 31, w = tid >> 5;
  __shared__ __bf16 pqs[128][72];   // [s][d], 144B rows (16B aligned)
  __shared__ __bf16 kvt[64][72];    // kv^T : [e][d]
  __shared__ float  ks[64];
  __shared__ float  den[128];
  const long long base = ((long long)b * Sh + s0) * Dh + h * 64;

  // stage pq tile (vectorized float4 -> packed bf16x4)
  #pragma unroll
  for (int c = 0; c < 8; ++c) {
    int id = tid + 256 * c;          // 0..2047 chunks of 4
    int s  = id >> 4;
    int dc = (id & 15) * 4;
    float4 f = *(const float4*)&pq[base + (long long)s * Dh + dc];
    v4bf q; q[0] = f2bf(f.x); q[1] = f2bf(f.y); q[2] = f2bf(f.z); q[3] = f2bf(f.w);
    *(v4bf*)&pqs[s][dc] = q;
  }
  // stage kv transposed
  for (int i = tid; i < 64 * 64; i += 256) {
    int r = i >> 6, c = i & 63;
    kvt[c][r] = f2bf(kv[(long long)bh * 4096 + i]);
  }
  if (tid < 64) ks[tid] = ksum[bh * 64 + tid];
  __syncthreads();
  if (tid < 128) {
    float s = 0.0f;
    for (int d = 0; d < 64; ++d) s += bf2f(pqs[tid][d]) * ks[d];
    den[tid] = s + 1e-6f;
  }
  __syncthreads();

  v8f acc[4];
  #pragma unroll
  for (int ni = 0; ni < 4; ++ni)
    #pragma unroll
    for (int r = 0; r < 8; ++r) acc[ni][r] = 0.0f;

  const int kg = lane >> 4, l15 = lane & 15;
  #pragma unroll
  for (int kk = 0; kk < 64; kk += 32) {
    int row = w * 16 + l15;
    v8bf alo = *(const v8bf*)&pqs[row][kk + kg * 8];
    v8bf ahi = *(const v8bf*)&pqs[row][kk + 16 + kg * 8];
    v16bf a = __builtin_bit_cast(v16bf, BF16x16{alo, ahi});
    #pragma unroll
    for (int ni = 0; ni < 4; ++ni) {
      int col = ni * 16 + l15;
      v8bf blo = *(const v8bf*)&kvt[col][kk + kg * 8];
      v8bf bhi = *(const v8bf*)&kvt[col][kk + 16 + kg * 8];
      v16bf bb = __builtin_bit_cast(v16bf, BF16x16{blo, bhi});
      acc[ni] = __builtin_amdgcn_wmma_f32_16x16x32_bf16(
          false, a, false, bb, (short)0, acc[ni], false, false);
    }
  }
  #pragma unroll
  for (int ni = 0; ni < 4; ++ni) {
    int n = ni * 16 + (lane & 15);
    #pragma unroll
    for (int r = 0; r < 8; ++r) {
      int m = w * 16 + (lane >> 4) * 8 + r;
      out[base + (long long)m * Dh + n] = acc[ni][r] / den[m];
    }
  }
}

// ---------------------------------------------------------------------------
// LayerNorm kernels (D=1024, block 256, 4 elems/thread, one block per row)
// ---------------------------------------------------------------------------
__device__ __forceinline__ float blk_sum(float v, float* red) {
  int tid = threadIdx.x;
  red[tid] = v; __syncthreads();
  for (int s = 128; s > 0; s >>= 1) { if (tid < s) red[tid] += red[tid + s]; __syncthreads(); }
  float r = red[0]; __syncthreads();
  return r;
}

__global__ __launch_bounds__(256)
void ln_kernel(const float* __restrict__ in, const float* __restrict__ g,
               const float* __restrict__ b, float* __restrict__ out)
{
  long long row = blockIdx.x;
  const float* p = in + row * Dh;
  float* o = out + row * Dh;
  __shared__ float red[256];
  int tid = threadIdx.x;
  float x[4], s = 0.0f;
  #pragma unroll
  for (int i = 0; i < 4; ++i) { x[i] = p[tid + 256 * i]; s += x[i]; }
  float mean = blk_sum(s, red) * (1.0f / Dh);
  float vs = 0.0f;
  #pragma unroll
  for (int i = 0; i < 4; ++i) { float d = x[i] - mean; vs += d * d; }
  float inv = rsqrtf(blk_sum(vs, red) * (1.0f / Dh) + 1e-5f);
  #pragma unroll
  for (int i = 0; i < 4; ++i) { int c = tid + 256 * i; o[c] = g[c] * (x[i] - mean) * inv + b[c]; }
}

// x1 = LN(in; g1,b1); h = LN(x1; g0,b0)
__global__ __launch_bounds__(256)
void fused_ln2(const float* __restrict__ in,
               const float* __restrict__ g1, const float* __restrict__ b1,
               const float* __restrict__ g0, const float* __restrict__ b0,
               float* __restrict__ x1, float* __restrict__ h)
{
  long long row = blockIdx.x;
  const float* p = in + row * Dh;
  __shared__ float red[256];
  int tid = threadIdx.x;
  float x[4], s = 0.0f;
  #pragma unroll
  for (int i = 0; i < 4; ++i) { x[i] = p[tid + 256 * i]; s += x[i]; }
  float mean = blk_sum(s, red) * (1.0f / Dh);
  float vs = 0.0f;
  #pragma unroll
  for (int i = 0; i < 4; ++i) { float d = x[i] - mean; vs += d * d; }
  float inv = rsqrtf(blk_sum(vs, red) * (1.0f / Dh) + 1e-5f);
  float y[4]; s = 0.0f;
  #pragma unroll
  for (int i = 0; i < 4; ++i) {
    int c = tid + 256 * i;
    y[i] = g1[c] * (x[i] - mean) * inv + b1[c];
    x1[row * Dh + c] = y[i];
    s += y[i];
  }
  float mean2 = blk_sum(s, red) * (1.0f / Dh);
  vs = 0.0f;
  #pragma unroll
  for (int i = 0; i < 4; ++i) { float d = y[i] - mean2; vs += d * d; }
  float inv2 = rsqrtf(blk_sum(vs, red) * (1.0f / Dh) + 1e-5f);
  #pragma unroll
  for (int i = 0; i < 4; ++i) { int c = tid + 256 * i; h[row * Dh + c] = g0[c] * (y[i] - mean2) * inv2 + b0[c]; }
}

// softmax over 64 slots * write_gate[row], in place; block=64, grid=B*T
__global__ __launch_bounds__(64)
void slot_softmax_gate(float* __restrict__ logits, const float* __restrict__ gate)
{
  int row = blockIdx.x, t = threadIdx.x;
  __shared__ float red[64];
  float* p = logits + (long long)row * 64;
  float v = p[t];
  red[t] = v; __syncthreads();
  for (int s = 32; s > 0; s >>= 1) { if (t < s) red[t] = fmaxf(red[t], red[t + s]); __syncthreads(); }
  float mx = red[0]; __syncthreads();
  float e = __expf(v - mx);
  red[t] = e; __syncthreads();
  for (int s = 32; s > 0; s >>= 1) { if (t < s) red[t] += red[t + s]; __syncthreads(); }
  p[t] = e / red[0] * gate[row];
}

// out_cache[b, LAYER*K + k, :] += update[b, k, :]
__global__ __launch_bounds__(256)
void cache_add(float* __restrict__ outc, const float* __restrict__ upd, int n)
{
  int i = blockIdx.x * 256 + threadIdx.x;
  if (i < n) {
    int b = i / (Kslt * DCh);
    int r = i % (Kslt * DCh);
    outc[(long long)b * (Lh * Kslt * DCh) + LAY * Kslt * DCh + r] += upd[i];
  }
}

// ---------------------------------------------------------------------------
// Host orchestration
// ---------------------------------------------------------------------------
static inline int cdiv(int a, int b) { return (a + b - 1) / b; }

extern "C" void kernel_launch(void* const* d_in, const int* in_sizes, int n_in,
                              void* d_out, int out_size, void* d_ws, size_t ws_size,
                              hipStream_t stream)
{
  (void)in_sizes; (void)n_in; (void)out_size; (void)ws_size;
  const float* x     = (const float*)d_in[0];
  const float* cache = (const float*)d_in[1];
  const float* Wq_r  = (const float*)d_in[2];
  const float* Wo_r  = (const float*)d_in[3];
  const float* Wg_r  = (const float*)d_in[4];
  const float* bg_r  = (const float*)d_in[5];
  const float* Wq    = (const float*)d_in[6];
  const float* Wk    = (const float*)d_in[7];
  const float* Wv    = (const float*)d_in[8];
  const float* Wo    = (const float*)d_in[9];
  const float* ln0_g = (const float*)d_in[10];
  const float* ln0_b = (const float*)d_in[11];
  const float* W1    = (const float*)d_in[12];
  const float* b1    = (const float*)d_in[13];
  const float* W2    = (const float*)d_in[14];
  const float* b2    = (const float*)d_in[15];
  const float* n1_g  = (const float*)d_in[16];
  const float* n1_b  = (const float*)d_in[17];
  const float* n2_g  = (const float*)d_in[18];
  const float* n2_b  = (const float*)d_in[19];
  /* d_in[20] = Ws : computed-but-unused in reference */
  const float* Wgw   = (const float*)d_in[21];
  const float* bgw   = (const float*)d_in[22];
  const float* Wslot = (const float*)d_in[23];
  const float* Wvw   = (const float*)d_in[24];

  float* W  = (float*)d_ws;
  float* Y  = (float*)d_out;          // y: (B,S,D)
  float* OC = Y + BSD;                // new_cache: (B, L*K, DC)

  const int BS = Bh * Sh;             // 16384 rows
  const float SC = 0.04419417382415922f;  // 1/sqrt(512)

  // workspace layout (floats), lifetime-based reuse
  const long long OFF_XF   = 0;                 // x_fused -> ffnsum
  const long long OFF_PQ   = BSD;               // pq -> h -> write-stage
  const long long OFF_PK   = 2 * BSD;           // scores -> pk -> sum1
  const long long OFF_V    = 3 * BSD;           // ctx -> v -> x1
  const long long OFF_KV   = 4 * BSD;           // gate -> kv
  const long long OFF_KS   = OFF_KV + 262144;   // ksum
  const long long OFF_AP   = OFF_KS + 4096;     // q_r -> attn_pre -> ffn hidden(bf16)
  const long long OFF_QR   = OFF_AP;
  const long long OFF_SC_  = OFF_PK;
  const long long OFF_CTX  = OFF_V;
  const long long OFF_G    = OFF_KV;
  const long long OFF_SUM1 = OFF_PK;
  const long long OFF_X1   = OFF_V;
  const long long OFF_H    = OFF_PQ;
  const long long OFF_T    = OFF_AP;
  const long long OFF_FS   = OFF_XF;
  const long long OFF_S2   = OFF_PQ;
  const long long OFF_VALS = OFF_PQ + 600000;
  const long long OFF_LG   = OFF_PQ + 1200000;
  const long long OFF_GW   = OFF_PQ + 1300000;
  const long long OFF_UPD  = OFF_PQ + 1310000;

  #define GL(TA,TB,ABF,CBF,EPI, A,lda,sa, Bp,ldb,sb, C,ldc,sc, M,N,Kd, bias,gate,sg, add, scl, Z) \
    gemm_wmma<TA,TB,ABF,CBF,EPI><<<dim3(cdiv(N,128),cdiv(M,128),Z),256,0,stream>>>( \
      A,lda,sa, Bp,ldb,sb, C,ldc,sc, M,N,Kd, bias,gate,sg, add, scl)

  // 1) q_r = x @ Wq_r
  GL(false,false,false,false,0, x,Dh,0, Wq_r,DCh,0, W+OFF_QR,DCh,0,
     BS,DCh,Dh, nullptr,nullptr,0, nullptr, 1.0f, 1);
  // 2) scores = q_r @ cache^T / sqrt(DC)
  GL(false,true,false,false,0, W+OFF_QR,DCh,(long long)Sh*DCh, cache,DCh,(long long)MS*DCh,
     W+OFF_SC_,MS,(long long)Sh*MS, Sh,MS,DCh, nullptr,nullptr,0, nullptr, SC, Bh);
  // 3) softmax over 768 slots
  softmax_rows<<<BS, 256, 0, stream>>>(W + OFF_SC_, MS);
  // 4) ctx = attn @ cache
  GL(false,false,false,false,0, W+OFF_SC_,MS,(long long)Sh*MS, cache,DCh,(long long)MS*DCh,
     W+OFF_CTX,DCh,(long long)Sh*DCh, Sh,DCh,MS, nullptr,nullptr,0, nullptr, 1.0f, Bh);
  // 5) read_gate = sigmoid(x @ Wg_r + bg_r)
  rowdot_sigmoid<<<dim3(cdiv(BS,8),1), 256, 0, stream>>>(x, 0, BS, Dh, Wg_r, bg_r, W+OFF_G, 0);
  // 6) x_fused = x + gate * (ctx @ Wo_r)
  GL(false,false,false,false,5, W+OFF_CTX,DCh,0, Wo_r,Dh,0, W+OFF_XF,Dh,0,
     BS,Dh,DCh, nullptr, W+OFF_G, 0, x, 1.0f, 1);
  // 7) pq = elu(xf@Wq)+1 ; pk = elu(xf@Wk)+1 ; v = xf@Wv
  GL(false,false,false,false,3, W+OFF_XF,Dh,0, Wq,Dh,0, W+OFF_PQ,Dh,0,
     BS,Dh,Dh, nullptr,nullptr,0, nullptr, 1.0f, 1);
  GL(false,false,false,false,3, W+OFF_XF,Dh,0, Wk,Dh,0, W+OFF_PK,Dh,0,
     BS,Dh,Dh, nullptr,nullptr,0, nullptr, 1.0f, 1);
  GL(false,false,false,false,0, W+OFF_XF,Dh,0, Wv,Dh,0, W+OFF_V,Dh,0,
     BS,Dh,Dh, nullptr,nullptr,0, nullptr, 1.0f, 1);
  // 8/9) kv + ksum
  linattn_kv<<<Bh*Hh, 256, 0, stream>>>(W+OFF_PK, W+OFF_V, W+OFF_KV);
  ksum_kernel<<<Bh*Hh, 64, 0, stream>>>(W+OFF_PK, W+OFF_KS);
  // 10) attn_pre
  linattn_num<<<dim3(Sh/128, Bh*Hh), 256, 0, stream>>>(W+OFF_PQ, W+OFF_KV, W+OFF_KS, W+OFF_AP);
  // 11) sum1 = x_fused + attn_pre @ Wo
  GL(false,false,false,false,4, W+OFF_AP,Dh,0, Wo,Dh,0, W+OFF_SUM1,Dh,0,
     BS,Dh,Dh, nullptr,nullptr,0, W+OFF_XF, 1.0f, 1);
  // 12) x1 = LN(sum1; n1), h = LN(x1; ln0)
  fused_ln2<<<BS, 256, 0, stream>>>(W+OFF_SUM1, n1_g, n1_b, ln0_g, ln0_b, W+OFF_X1, W+OFF_H);
  // 13) t = gelu(h @ W1 + b1) -> bf16
  GL(false,false,false,true,2, W+OFF_H,Dh,0, W1,4*Dh,0, W+OFF_T,4*Dh,0,
     BS,4*Dh,Dh, b1,nullptr,0, nullptr, 1.0f, 1);
  // 14) ffnsum = x1 + t @ W2 + b2
  GL(false,false,true,false,6, W+OFF_T,4*Dh,0, W2,Dh,0, W+OFF_FS,Dh,0,
     BS,Dh,4*Dh, b2,nullptr,0, W+OFF_X1, 1.0f, 1);
  // 15) y = LN(ffnsum; n2) -> d_out
  ln_kernel<<<BS, 256, 0, stream>>>(W+OFF_FS, n2_g, n2_b, Y);
  // 16) new_cache = copy(cache)
  hipMemcpyAsync(OC, cache, (size_t)Bh * Lh * Kslt * DCh * sizeof(float),
                 hipMemcpyDeviceToDevice, stream);
  // 17) s2 = wc @ Wslot
  GL(false,false,false,false,0, Y,Dh,(long long)Sh*Dh, Wslot,DCh,0,
     W+OFF_S2,DCh,(long long)Th*DCh, Th,DCh,Dh, nullptr,nullptr,0, nullptr, 1.0f, Bh);
  // 18) slot_logits = s2 @ local^T / sqrt(DC)
  GL(false,true,false,false,0, W+OFF_S2,DCh,(long long)Th*DCh,
     cache + (long long)LAY*Kslt*DCh, DCh, (long long)Lh*Kslt*DCh,
     W+OFF_LG,Kslt,(long long)Th*Kslt, Th,Kslt,DCh, nullptr,nullptr,0, nullptr, SC, Bh);
  // 19) write_gate = sigmoid(wc @ Wgw + bgw)
  rowdot_sigmoid<<<dim3(cdiv(Th,8),Bh), 256, 0, stream>>>(Y, (long long)Sh*Dh, Th, Dh,
                                                          Wgw, bgw, W+OFF_GW, Th);
  // 20) probs = softmax(logits) * gate (in place)
  slot_softmax_gate<<<Bh*Th, 64, 0, stream>>>(W+OFF_LG, W+OFF_GW);
  // 21) vals = wc @ Wvw
  GL(false,false,false,false,0, Y,Dh,(long long)Sh*Dh, Wvw,DCh,0,
     W+OFF_VALS,DCh,(long long)Th*DCh, Th,DCh,Dh, nullptr,nullptr,0, nullptr, 1.0f, Bh);
  // 22) update = probs^T @ vals
  GL(true,false,false,false,0, W+OFF_LG,Kslt,(long long)Th*Kslt,
     W+OFF_VALS,DCh,(long long)Th*DCh,
     W+OFF_UPD,DCh,(long long)Kslt*DCh, Kslt,DCh,Th, nullptr,nullptr,0, nullptr, 1.0f, Bh);
  // 23) new_cache[:, LAYER*K:(LAYER+1)*K, :] += update
  cache_add<<<cdiv(Bh*Kslt*DCh,256), 256, 0, stream>>>(OC, W+OFF_UPD, Bh*Kslt*DCh);

  #undef GL
}
